// myshgn_23055384445695
// MI455X (gfx1250) — matrix operand order repaired and checked by reading, not compiled, not run on hardware
//
#include <hip/hip_runtime.h>
#include <hip/hip_bf16.h>
#include <math.h>

// ---------------- problem constants ----------------
#define NN0 50000
#define NN1 30000
#define NN2 20000
#define NTOT (NN0 + NN1 + NN2)   // 100000 (divisible by 16)
#define NE 1600000
#define HID 32
#define HEADS 4
#define EDIM 32
#define NET 5
#define SLOPE 0.2f
#define BETA 0.05f

// ---------------- WMMA types ----------------
typedef __bf16 bf16_t;
typedef bf16_t v16bf __attribute__((ext_vector_type(16)));
typedef float  v8f   __attribute__((ext_vector_type(8)));
typedef unsigned short v16us __attribute__((ext_vector_type(16)));
typedef unsigned short v8us  __attribute__((ext_vector_type(8)));

__device__ inline unsigned short f2bf(float f) {
    unsigned int u = __float_as_uint(f);
    unsigned int r = u + 0x7FFFu + ((u >> 16) & 1u);   // round-to-nearest-even
    return (unsigned short)(r >> 16);
}

__device__ inline void atomicMaxF(float* addr, float v) {
    if (v >= 0.f) atomicMax((int*)addr, __float_as_int(v));
    else          atomicMin((unsigned int*)addr, __float_as_uint(v));
}

// ---------------- generic fill ----------------
__global__ void k_fill(float* __restrict__ p, float v, int n) {
    int i = blockIdx.x * blockDim.x + threadIdx.x;
    if (i < n) p[i] = v;
}

// ---------------- pack weight (K x NC fp32 row-major) -> bf16 B fragments ----
__global__ void k_pack_b(const float* __restrict__ W, int K, int NC,
                         unsigned short* __restrict__ outp) {
    int t = blockIdx.x * blockDim.x + threadIdx.x;
    int KT = K >> 5, NT = NC >> 4;
    int total = KT * NT * 32;
    if (t >= total) return;
    int lane = t & 31;
    int frag = t >> 5;
    int nt = frag % NT, kt = frag / NT;
    int col = nt * 16 + (lane & 15);
    int kb  = kt * 32 + ((lane < 16) ? 0 : 8);
    unsigned short* o = outp + (size_t)t * 16;
#pragma unroll
    for (int i = 0; i < 16; ++i)
        o[i] = f2bf(W[(size_t)(kb + i + ((i >= 8) ? 8 : 0)) * NC + col]);
}

// ---------------- A-fragment loaders ----------------
// layout: lanes 0-15 hold K {0..7,16..23}; lanes 16-31 hold K {8..15,24..31}
__device__ inline v16bf load_a_frag(const float* ap) {
    union { v16bf v; unsigned short u[16]; } af;
#pragma unroll
    for (int i = 0; i < 8; ++i) af.u[i]     = f2bf(ap[i]);
#pragma unroll
    for (int i = 0; i < 8; ++i) af.u[8 + i] = f2bf(ap[16 + i]);
    return af.v;
}
__device__ inline v16bf load_a_frag(const unsigned short* ap) {
    union { v16bf v; v8us h[2]; } af;
    af.h[0] = *(const v8us*)(ap);        // 16B
    af.h[1] = *(const v8us*)(ap + 16);   // 16B
    return af.v;
}

// ---------------- output store ----------------
__device__ inline void store_c(float* C, size_t idx, float v) { C[idx] = v; }
__device__ inline void store_c(unsigned short* C, size_t idx, float v) { C[idx] = f2bf(v); }

// ---------------- WMMA GEMM ------------------------------------------------
// C[M x (NT*16)] = A[M x KT*32] * Bpack (+bias). One wave computes TWO adjacent
// 16x16 output tiles (shared A fragment). K fully unrolled at compile time.
template <typename AT, typename OT, int KT>
__global__ void k_gemm_wmma(const AT* __restrict__ A, int M,
                            const unsigned short* __restrict__ Bp, int NT,
                            OT* __restrict__ C, const float* __restrict__ bias) {
    const int K = KT * 32;
    int lane = threadIdx.x & 31;
    int wv   = threadIdx.x >> 5;
    int NT2  = NT >> 1;
    int tile = blockIdx.x * 8 + wv;
    int MT = M >> 4;                       // M always a multiple of 16
    if (tile >= MT * NT2) return;
    int mt = tile / NT2;
    int nt0 = (tile % NT2) * 2;

    int row = mt * 16 + (lane & 15);
    int klo = (lane < 16) ? 0 : 8;

    v8f acc0 = {}, acc1 = {};
#pragma unroll
    for (int kt = 0; kt < KT; ++kt) {
        v16bf av = load_a_frag(A + (size_t)row * K + kt * 32 + klo);
        const unsigned short* bp0 = Bp + ((size_t)(kt * NT + nt0) * 32 + lane) * 16;
        v16bf b0 = __builtin_bit_cast(v16bf, *(const v16us*)bp0);
        v16bf b1 = __builtin_bit_cast(v16bf, *(const v16us*)(bp0 + 32 * 16));
        acc0 = __builtin_amdgcn_wmma_f32_16x16x32_bf16(
            false, av, false, b0, (short)0, acc0, false, false);
        acc1 = __builtin_amdgcn_wmma_f32_16x16x32_bf16(
            false, av, false, b1, (short)0, acc1, false, false);
    }

    int NC  = NT * 16;
    int c0  = nt0 * 16 + (lane & 15);
    int r0  = mt * 16 + ((lane < 16) ? 0 : 8);
    float b0v = bias ? bias[c0]      : 0.f;
    float b1v = bias ? bias[c0 + 16] : 0.f;
#pragma unroll
    for (int j = 0; j < 8; ++j) {
        size_t base = (size_t)(r0 + j) * NC;
        store_c(C, base + c0,      acc0[j] + b0v);
        store_c(C, base + c0 + 16, acc1[j] + b1v);
    }
}

// ---------------- he[t,h] precompute ----------------
__global__ void k_prep_he(const float* __restrict__ ee, const float* __restrict__ Wr,
                          const float* __restrict__ ae, float* __restrict__ he) {
    __shared__ float te[NET * 128];
    int t = threadIdx.x;
    if (t < NET * 128) {
        int ty = t >> 7, o = t & 127;
        float s = 0.f;
        for (int e = 0; e < EDIM; ++e)
            s += ee[ty * EDIM + e] * Wr[((size_t)ty * EDIM + e) * 128 + o];
        te[t] = s;
    }
    __syncthreads();
    if (t < NET * HEADS) {
        int ty = t >> 2, hd = t & 3;
        float s = 0.f;
        for (int k = 0; k < EDIM; ++k)
            s += te[ty * 128 + hd * 32 + k] * ae[hd * 32 + k];
        he[t] = s;
    }
}

// ---------------- hl/hr per (node, head) ----------------
__global__ void k_attn_coef(const float* __restrict__ emb, const float* __restrict__ al,
                            const float* __restrict__ ar, float* __restrict__ hl,
                            float* __restrict__ hr, int n4) {
    int idx = blockIdx.x * blockDim.x + threadIdx.x;
    if (idx >= n4) return;
    int hd = idx & 3;
    const float* ep = emb + (size_t)(idx >> 2) * 128 + hd * 32;
    float sl = 0.f, sr = 0.f;
#pragma unroll
    for (int k = 0; k < 32; ++k) {
        float e = ep[k];
        sl += e * al[hd * 32 + k];
        sr += e * ar[hd * 32 + k];
    }
    hl[idx] = sl; hr[idx] = sr;
}

// ---------------- edge logits + segment max ----------------
__global__ void k_edge_logit_max(const int* __restrict__ src, const int* __restrict__ dst,
                                 const int* __restrict__ ety, const float* __restrict__ hl,
                                 const float* __restrict__ hr, const float* __restrict__ he,
                                 float* __restrict__ logit, float* __restrict__ m) {
    int idx = blockIdx.x * blockDim.x + threadIdx.x;
    if (idx >= NE * HEADS) return;
    int e = idx >> 2, hd = idx & 3;
    int s = src[e], d = dst[e], t = ety[e];
    float z = hl[s * 4 + hd] + hr[d * 4 + hd] + he[t * 4 + hd];
    z = (z > 0.f) ? z : SLOPE * z;
    logit[idx] = z;
    atomicMaxF(&m[d * 4 + hd], z);
}

// ---------------- exp + segment sum (in place) ----------------
__global__ void k_exp_sum(const int* __restrict__ dst, float* __restrict__ logit,
                          const float* __restrict__ m, float* __restrict__ den) {
    int idx = blockIdx.x * blockDim.x + threadIdx.x;
    if (idx >= NE * HEADS) return;
    int e = idx >> 2, hd = idx & 3;
    int d = dst[e];
    float ex = __expf(logit[idx] - m[d * 4 + hd]);
    logit[idx] = ex;
    atomicAdd(&den[d * 4 + hd], ex);
}

// ---------------- layer0: att (stored) + message scatter-add ----------------
__global__ void k_att_agg0(const int* __restrict__ src, const int* __restrict__ dst,
                           const float* __restrict__ ex, const float* __restrict__ den,
                           const float* __restrict__ emb, float* __restrict__ att0,
                           float* __restrict__ agg) {
    int idx = blockIdx.x * blockDim.x + threadIdx.x;
    if (idx >= NE * HEADS) return;
    int e = idx >> 2, hd = idx & 3;
    int d = dst[e], s = src[e];
    float att = ex[idx] / den[d * 4 + hd];
    att0[idx] = att;
    const float* ep = emb + (size_t)s * 128 + hd * 32;
    float* op = agg + (size_t)d * 128 + hd * 32;
#pragma unroll
    for (int k = 0; k < 32; ++k) atomicAdd(&op[k], ep[k] * att);
}

// ---------------- layer1: blended att + scatter into out (n, k*4+hd) --------
__global__ void k_att_agg1(const int* __restrict__ src, const int* __restrict__ dst,
                           const float* __restrict__ ex, const float* __restrict__ den,
                           const float* __restrict__ att0, const float* __restrict__ emb,
                           float* __restrict__ outp) {
    int idx = blockIdx.x * blockDim.x + threadIdx.x;
    if (idx >= NE * HEADS) return;
    int e = idx >> 2, hd = idx & 3;
    int d = dst[e], s = src[e];
    float att = (ex[idx] / den[d * 4 + hd]) * (1.f - BETA) + att0[idx] * BETA;
    const float* ep = emb + (size_t)s * 128 + hd * 32;
    float* op = outp + (size_t)d * 128;
#pragma unroll
    for (int k = 0; k < 32; ++k) atomicAdd(&op[k * 4 + hd], ep[k] * att);
}

// ---------------- transpose (n,h,d)->(n,d*4+h) + elu -> bf16 ----------------
__global__ void k_trans_elu(const float* __restrict__ agg, unsigned short* __restrict__ h1) {
    int idx = blockIdx.x * blockDim.x + threadIdx.x;
    if (idx >= NTOT * 128) return;
    int n = idx >> 7, r = idx & 127, hd = r >> 5, k = r & 31;
    float v = agg[idx];
    float o = (v > 0.f) ? v : (__expf(v) - 1.f);
    h1[(size_t)n * 128 + k * 4 + hd] = f2bf(o);
}

// ======================= host side =======================
extern "C" void kernel_launch(void* const* d_in, const int* in_sizes, int n_in,
                              void* d_out, int out_size, void* d_ws, size_t ws_size,
                              hipStream_t stream) {
    const float* x0    = (const float*)d_in[0];
    const float* x1    = (const float*)d_in[1];
    const float* x2    = (const float*)d_in[2];
    const float* fc0_w = (const float*)d_in[3];
    const float* fc0_b = (const float*)d_in[4];
    const float* fc1_w = (const float*)d_in[5];
    const float* fc1_b = (const float*)d_in[6];
    const float* fc2_w = (const float*)d_in[7];
    const float* fc2_b = (const float*)d_in[8];
    const float* l0_W  = (const float*)d_in[9];
    const float* l0_ee = (const float*)d_in[10];
    const float* l0_Wr = (const float*)d_in[11];
    const float* l0_al = (const float*)d_in[12];
    const float* l0_ar = (const float*)d_in[13];
    const float* l0_ae = (const float*)d_in[14];
    const float* l1_W  = (const float*)d_in[15];
    const float* l1_ee = (const float*)d_in[16];
    const float* l1_Wr = (const float*)d_in[17];
    const float* l1_al = (const float*)d_in[18];
    const float* l1_ar = (const float*)d_in[19];
    const float* l1_ae = (const float*)d_in[20];
    const float* l1_rw = (const float*)d_in[21];
    const float* l1_rb = (const float*)d_in[22];
    const int*   srcI  = (const int*)d_in[23];
    const int*   dstI  = (const int*)d_in[24];
    const int*   etyI  = (const int*)d_in[25];
    float* outp = (float*)d_out;

    // ---- carve workspace ----
    char* ws = (char*)d_ws;
    size_t off = 0;
    auto alloc = [&](size_t bytes) -> void* {
        off = (off + 255) & ~(size_t)255;
        void* p = ws + off;
        off += bytes;
        return p;
    };
    float* he0  = (float*)alloc(NET * HEADS * 4);
    float* he1  = (float*)alloc(NET * HEADS * 4);
    unsigned short* pfc0 = (unsigned short*)alloc((size_t)8 * 2 * 32 * 16 * 2);
    unsigned short* pfc1 = (unsigned short*)alloc((size_t)4 * 2 * 32 * 16 * 2);
    unsigned short* pfc2 = (unsigned short*)alloc((size_t)2 * 2 * 32 * 16 * 2);
    unsigned short* pl0W = (unsigned short*)alloc((size_t)1 * 8 * 32 * 16 * 2);
    unsigned short* pl1W = (unsigned short*)alloc((size_t)4 * 8 * 32 * 16 * 2);
    unsigned short* plrw = (unsigned short*)alloc((size_t)4 * 8 * 32 * 16 * 2);
    unsigned short* h0   = (unsigned short*)alloc((size_t)NTOT * 32 * 2);   // bf16
    float* emb  = (float*)alloc((size_t)NTOT * 128 * 4);    // emb0, reused as emb1
    float* hl   = (float*)alloc((size_t)NTOT * 4 * 4);
    float* hr   = (float*)alloc((size_t)NTOT * 4 * 4);
    float* mbuf = (float*)alloc((size_t)NTOT * 4 * 4);
    float* den  = (float*)alloc((size_t)NTOT * 4 * 4);
    float* logit= (float*)alloc((size_t)NE * 4 * 4);
    float* att0 = (float*)alloc((size_t)NE * 4 * 4);
    float* agg  = (float*)alloc((size_t)NTOT * 128 * 4);
    unsigned short* h1 = (unsigned short*)alloc((size_t)NTOT * 128 * 2);    // bf16

    const int T = 256;
    auto blks = [](long long n, int t) { return (int)((n + t - 1) / t); };
    auto gblk = [](int M, int NT) { return ((M / 16) * (NT / 2) + 7) / 8; };

    // ---- weight packing + he precompute ----
    k_pack_b<<<blks(8 * 2 * 32, T), T, 0, stream>>>(fc0_w, 256, 32, pfc0);
    k_pack_b<<<blks(4 * 2 * 32, T), T, 0, stream>>>(fc1_w, 128, 32, pfc1);
    k_pack_b<<<blks(2 * 2 * 32, T), T, 0, stream>>>(fc2_w, 64, 32, pfc2);
    k_pack_b<<<blks(1 * 8 * 32, T), T, 0, stream>>>(l0_W, 32, 128, pl0W);
    k_pack_b<<<blks(4 * 8 * 32, T), T, 0, stream>>>(l1_W, 128, 128, pl1W);
    k_pack_b<<<blks(4 * 8 * 32, T), T, 0, stream>>>(l1_rw, 128, 128, plrw);
    k_prep_he<<<1, 640, 0, stream>>>(l0_ee, l0_Wr, l0_ae, he0);
    k_prep_he<<<1, 640, 0, stream>>>(l1_ee, l1_Wr, l1_ae, he1);

    // ---- input projections (fp32 A -> bf16 h0) ----
    k_gemm_wmma<float, unsigned short, 8><<<gblk(NN0, 2), T, 0, stream>>>(
        x0, NN0, pfc0, 2, h0, fc0_b);
    k_gemm_wmma<float, unsigned short, 4><<<gblk(NN1, 2), T, 0, stream>>>(
        x1, NN1, pfc1, 2, h0 + (size_t)NN0 * 32, fc1_b);
    k_gemm_wmma<float, unsigned short, 2><<<gblk(NN2, 2), T, 0, stream>>>(
        x2, NN2, pfc2, 2, h0 + (size_t)(NN0 + NN1) * 32, fc2_b);

    // ================= layer 0 =================
    k_gemm_wmma<unsigned short, float, 1><<<gblk(NTOT, 8), T, 0, stream>>>(
        h0, NTOT, pl0W, 8, emb, nullptr);                            // emb0
    k_attn_coef<<<blks((long long)NTOT * 4, T), T, 0, stream>>>(emb, l0_al, l0_ar, hl, hr, NTOT * 4);
    k_fill<<<blks((long long)NTOT * 4, T), T, 0, stream>>>(mbuf, -__builtin_inff(), NTOT * 4);
    k_fill<<<blks((long long)NTOT * 4, T), T, 0, stream>>>(den, 0.f, NTOT * 4);
    k_fill<<<blks((long long)NTOT * 128, T), T, 0, stream>>>(agg, 0.f, NTOT * 128);
    k_edge_logit_max<<<blks((long long)NE * 4, T), T, 0, stream>>>(srcI, dstI, etyI, hl, hr, he0, logit, mbuf);
    k_exp_sum<<<blks((long long)NE * 4, T), T, 0, stream>>>(dstI, logit, mbuf, den);
    k_att_agg0<<<blks((long long)NE * 4, T), T, 0, stream>>>(srcI, dstI, logit, den, emb, att0, agg);
    k_trans_elu<<<blks((long long)NTOT * 128, T), T, 0, stream>>>(agg, h1);

    // ================= layer 1 =================
    k_gemm_wmma<unsigned short, float, 4><<<gblk(NTOT, 8), T, 0, stream>>>(
        h1, NTOT, pl1W, 8, emb, nullptr);                            // emb1
    k_attn_coef<<<blks((long long)NTOT * 4, T), T, 0, stream>>>(emb, l1_al, l1_ar, hl, hr, NTOT * 4);
    k_fill<<<blks((long long)NTOT * 4, T), T, 0, stream>>>(mbuf, -__builtin_inff(), NTOT * 4);
    k_fill<<<blks((long long)NTOT * 4, T), T, 0, stream>>>(den, 0.f, NTOT * 4);
    k_edge_logit_max<<<blks((long long)NE * 4, T), T, 0, stream>>>(srcI, dstI, etyI, hl, hr, he1, logit, mbuf);
    k_exp_sum<<<blks((long long)NE * 4, T), T, 0, stream>>>(dstI, logit, mbuf, den);
    // residual base: out = h1 @ res_w + res_b (completes before scatter-add)
    k_gemm_wmma<unsigned short, float, 4><<<gblk(NTOT, 8), T, 0, stream>>>(
        h1, NTOT, plrw, 8, outp, l1_rb);
    // blended attention message aggregation directly into out (n, d*4+h)
    k_att_agg1<<<blks((long long)NE * 4, T), T, 0, stream>>>(srcI, dstI, logit, den, att0, emb, outp);
}
